// Encoder_Decoder_Model_Homo_75256416960673
// MI455X (gfx1250) — matrix operand order repaired and checked by reading, not compiled, not run on hardware
//
#include <hip/hip_runtime.h>

#define N_NODES 100000
#define D_IN    64
#define HID     128
#define N_EDGES 1600000
#define N_LBL   500000

typedef __attribute__((ext_vector_type(16))) _Float16 v16h;
typedef __attribute__((ext_vector_type(8)))  float    v8f;

union Frag16 { v16h h; unsigned int u[8]; };

// ---------------------------------------------------------------- utilities
__global__ void zero_f32_kernel(float* p, long long n) {
  long long i = (long long)blockIdx.x * blockDim.x + threadIdx.x;
  if (i < n) p[i] = 0.0f;
}

__global__ void degree_kernel(const long long* ei, float* deg) {
  int e = blockIdx.x * blockDim.x + threadIdx.x;
  if (e < N_EDGES) {
    int d = (int)ei[N_EDGES + e];           // row 1 = dst
    atomicAdd(&deg[d], 1.0f);
  }
}

__global__ void invdeg_kernel(const float* deg, float* invd) {
  int i = blockIdx.x * blockDim.x + threadIdx.x;
  if (i < N_NODES) {
    float d = deg[i];
    invd[i] = (d > 0.0f) ? (1.0f / d) : 0.0f;
  }
}

// scatter-add feat[src] into agg[dst]; one thread = one edge x 4 floats
__global__ void scatter_add_kernel(const float* feat, const long long* ei,
                                   float* agg, int D) {
  long long t = (long long)blockIdx.x * blockDim.x + threadIdx.x;
  int chunks = D >> 2;
  long long total = (long long)N_EDGES * chunks;
  if (t >= total) return;
  int e = (int)(t / chunks);
  int c = (int)(t % chunks) << 2;
  int s = (int)ei[e];
  int d = (int)ei[N_EDGES + e];
  const float4 v = *(const float4*)(feat + (long long)s * D + c);
  float* dst = agg + (long long)d * D + c;
  atomicAdd(dst + 0, v.x);
  atomicAdd(dst + 1, v.y);
  atomicAdd(dst + 2, v.z);
  atomicAdd(dst + 3, v.w);
}

// Pack W = [wl ; wr] (K-major, N=HID cols) into f16 WMMA B-fragment order:
// out[((kt*8 + nt)*8 + v)*32 + lane], lane gives n & K-halfgroup, v gives K pair.
// B order used here (16b, 32x16; contiguous K, same as A): lanes 0-15 -> N=lane, K=2v,2v+1; lanes 16-31 -> K+16.
__global__ void pack_weights_kernel(const float* wl, const float* wr,
                                    int Din, int Ktot, unsigned* out) {
  int idx = blockIdx.x * blockDim.x + threadIdx.x;
  int total = (Ktot / 32) * 8 * 8 * 32;
  if (idx >= total) return;
  int lane = idx & 31;
  int v    = (idx >> 5) & 7;
  int nt   = (idx >> 8) & 7;
  int kt   = idx >> 11;
  int n  = nt * 16 + (lane & 15);
  int k0 = kt * 32 + ((lane >> 4) << 4) + 2 * v;
  int k1 = k0 + 1;
  float f0 = (k0 < Din) ? wl[(long long)k0 * HID + n]
                        : wr[(long long)(k0 - Din) * HID + n];
  float f1 = (k1 < Din) ? wl[(long long)k1 * HID + n]
                        : wr[(long long)(k1 - Din) * HID + n];
  _Float16 h0 = (_Float16)f0, h1 = (_Float16)f1;
  unsigned short s0, s1;
  __builtin_memcpy(&s0, &h0, 2);
  __builtin_memcpy(&s1, &h1, 2);
  out[idx] = (unsigned)s0 | ((unsigned)s1 << 16);
}

// ------------------------------------------------- fused SAGE linear layer
// Each wave owns TWO 16-node M-tiles (32 rows): every B fragment load feeds
// two back-to-back WMMAs (halves B bandwidth per wmma, doubles intensity).
template <int DIN>
__launch_bounds__(128)
__global__ void sage_linear_kernel(const float* __restrict__ agg,
                                   const float* __restrict__ invd,
                                   const float* __restrict__ x,
                                   const unsigned* __restrict__ packW,
                                   const float* __restrict__ bias,
                                   float* __restrict__ out, int do_relu) {
  constexpr int K2 = 2 * DIN;       // concat-K
  constexpr int KT = K2 / 32;       // WMMA K-steps
  __shared__ _Float16 smem[4 * 32 * K2];
  const int wave = threadIdx.x >> 5;
  const int lane = threadIdx.x & 31;
  const int pair = blockIdx.x * 4 + wave;   // covers rows [pair*32, pair*32+32)
  const int base = pair * 32;
  const bool active = base < N_NODES;       // 6250 tiles -> 3125 full pairs
  _Float16* my = smem + wave * 32 * K2;

  if (active) {
    for (int idx = lane; idx < 32 * K2; idx += 32) {
      const int r = idx / K2, c = idx % K2;
      const int node = base + r;
      float fv;
      if (c < DIN) fv = agg[(long long)node * DIN + c] * invd[node];
      else         fv = x[(long long)node * DIN + (c - DIN)];
      my[idx] = (_Float16)fv;
    }
  }
  __syncthreads();
  if (!active) return;

  const unsigned* sm32 = (const unsigned*)my;
  const int m = lane & 15;
  const int g = lane >> 4;

  for (int nt = 0; nt < 8; ++nt) {
    v8f acc0 = {}, acc1 = {};
#pragma unroll
    for (int kt = 0; kt < KT; ++kt) {
      Frag16 a0, a1, b;
#pragma unroll
      for (int v = 0; v < 8; ++v) {
        // A layout matching packed B: K=16g+2v,2v+1
        const int k = kt * 32 + g * 16 + 2 * v;
        a0.u[v] = sm32[(m * K2 + k) >> 1];
        a1.u[v] = sm32[((16 + m) * K2 + k) >> 1];
        b.u[v]  = packW[(((kt * 8 + nt) * 8 + v) << 5) + lane];
      }
      acc0 = __builtin_amdgcn_wmma_f32_16x16x32_f16(
          false, a0.h, false, b.h, (short)0, acc0, false, false);
      acc1 = __builtin_amdgcn_wmma_f32_16x16x32_f16(
          false, a1.h, false, b.h, (short)0, acc1, false, false);
    }
    const int n = nt * 16 + m;
    const float bv = bias[n];
#pragma unroll
    for (int r = 0; r < 8; ++r) {   // C/D: row = r + 8g, col = n
      float v0 = acc0[r] + bv;
      float v1 = acc1[r] + bv;
      if (do_relu) { v0 = fmaxf(v0, 0.0f); v1 = fmaxf(v1, 0.0f); }
      out[(long long)(base +      8 * g + r) * HID + n] = v0;
      out[(long long)(base + 16 + 8 * g + r) * HID + n] = v1;
    }
  }
}

// ------------------------------------------------------------ edge decoder
// one wave per 32 label-edges (two 16-row tiles); A = concat(z[s], z[d])
// (K=256); second matmul (128 -> 1) fused as per-lane dot + width-16
// xor-shuffle reduce.
__launch_bounds__(128)
__global__ void decoder_kernel(const float* __restrict__ z,
                               const long long* __restrict__ eli,
                               const unsigned* __restrict__ packW,
                               const float* __restrict__ db1,
                               const float* __restrict__ dw2,
                               const float* __restrict__ db2,
                               float* __restrict__ out) {
  constexpr int K2 = 256, KT = 8;
  __shared__ _Float16 smem[4 * 32 * K2];
  const int wave = threadIdx.x >> 5;
  const int lane = threadIdx.x & 31;
  const int pair = blockIdx.x * 4 + wave;
  const int base = pair * 32;
  const bool active = base < N_LBL;         // 31250 tiles -> 15625 full pairs
  _Float16* my = smem + wave * 32 * K2;

  if (active) {
    for (int idx = lane; idx < 32 * K2; idx += 32) {
      const int r = idx >> 8, c = idx & 255;
      const int e = base + r;
      const long long node = (c < HID) ? eli[e] : eli[N_LBL + e];
      const int cc = (c < HID) ? c : (c - HID);
      my[idx] = (_Float16)z[node * HID + cc];
    }
  }
  __syncthreads();
  if (!active) return;

  const unsigned* sm32 = (const unsigned*)my;
  const int m = lane & 15, g = lane >> 4;
  float dot0[8] = {0.f, 0.f, 0.f, 0.f, 0.f, 0.f, 0.f, 0.f};
  float dot1[8] = {0.f, 0.f, 0.f, 0.f, 0.f, 0.f, 0.f, 0.f};

  for (int nt = 0; nt < 8; ++nt) {
    v8f acc0 = {}, acc1 = {};
#pragma unroll
    for (int kt = 0; kt < KT; ++kt) {
      Frag16 a0, a1, b;
#pragma unroll
      for (int v = 0; v < 8; ++v) {
        const int k = kt * 32 + g * 16 + 2 * v;
        a0.u[v] = sm32[(m * K2 + k) >> 1];
        a1.u[v] = sm32[((16 + m) * K2 + k) >> 1];
        b.u[v]  = packW[(((kt * 8 + nt) * 8 + v) << 5) + lane];
      }
      acc0 = __builtin_amdgcn_wmma_f32_16x16x32_f16(
          false, a0.h, false, b.h, (short)0, acc0, false, false);
      acc1 = __builtin_amdgcn_wmma_f32_16x16x32_f16(
          false, a1.h, false, b.h, (short)0, acc1, false, false);
    }
    const int n = nt * 16 + m;
    const float bv = db1[n];
    const float wv = dw2[n];
#pragma unroll
    for (int r = 0; r < 8; ++r) {
      dot0[r] += fmaxf(acc0[r] + bv, 0.0f) * wv;
      dot1[r] += fmaxf(acc1[r] + bv, 0.0f) * wv;
    }
  }
  const float bias2 = db2[0];
#pragma unroll
  for (int r = 0; r < 8; ++r) {
    float s0 = dot0[r], s1 = dot1[r];
    s0 += __shfl_xor(s0, 1, 16); s1 += __shfl_xor(s1, 1, 16);
    s0 += __shfl_xor(s0, 2, 16); s1 += __shfl_xor(s1, 2, 16);
    s0 += __shfl_xor(s0, 4, 16); s1 += __shfl_xor(s1, 4, 16);
    s0 += __shfl_xor(s0, 8, 16); s1 += __shfl_xor(s1, 8, 16);
    if (m == 0) {
      out[base +      8 * g + r] = s0 + bias2;
      out[base + 16 + 8 * g + r] = s1 + bias2;
    }
  }
}

// --------------------------------------------------------------- launcher
extern "C" void kernel_launch(void* const* d_in, const int* in_sizes, int n_in,
                              void* d_out, int out_size, void* d_ws, size_t ws_size,
                              hipStream_t stream) {
  const float*     x   = (const float*)d_in[0];
  const long long* ei  = (const long long*)d_in[1];
  const long long* eli = (const long long*)d_in[2];
  const float*     w1l = (const float*)d_in[3];
  const float*     b1l = (const float*)d_in[4];
  const float*     w1r = (const float*)d_in[5];
  const float*     w2l = (const float*)d_in[6];
  const float*     b2l = (const float*)d_in[7];
  const float*     w2r = (const float*)d_in[8];
  const float*     dw1 = (const float*)d_in[9];
  const float*     db1 = (const float*)d_in[10];
  const float*     dw2 = (const float*)d_in[11];
  const float*     db2 = (const float*)d_in[12];
  float* out = (float*)d_out;

  // workspace layout (deg, agg1, agg2 contiguous so one zero pass covers them)
  float* ws   = (float*)d_ws;
  float* deg  = ws;
  float* agg1 = deg  + N_NODES;
  float* agg2 = agg1 + (size_t)N_NODES * 64;
  float* invd = agg2 + (size_t)N_NODES * 128;
  float* h    = invd + N_NODES;
  float* z    = h    + (size_t)N_NODES * 128;
  unsigned* packW1 = (unsigned*)(z + (size_t)N_NODES * 128);
  unsigned* packW2 = packW1 + 4 * 8 * 8 * 32;
  unsigned* packD1 = packW2 + 8 * 8 * 8 * 32;

  const long long nzero = (long long)N_NODES * (1 + 64 + 128);
  zero_f32_kernel<<<(unsigned)((nzero + 255) / 256), 256, 0, stream>>>(deg, nzero);

  degree_kernel<<<(N_EDGES + 255) / 256, 256, 0, stream>>>(ei, deg);
  invdeg_kernel<<<(N_NODES + 255) / 256, 256, 0, stream>>>(deg, invd);

  pack_weights_kernel<<<(4 * 8 * 8 * 32) / 256, 256, 0, stream>>>(w1l, w1r, 64, 128, packW1);
  pack_weights_kernel<<<(8 * 8 * 8 * 32) / 256, 256, 0, stream>>>(w2l, w2r, 128, 256, packW2);
  pack_weights_kernel<<<(8 * 8 * 8 * 32) / 256, 256, 0, stream>>>(dw1, dw1, 256, 256, packD1);

  {
    long long tot = (long long)N_EDGES * (64 / 4);
    scatter_add_kernel<<<(unsigned)((tot + 255) / 256), 256, 0, stream>>>(x, ei, agg1, 64);
  }
  {
    int pairs = (N_NODES + 31) / 32;    // 3125
    sage_linear_kernel<64><<<(pairs + 3) / 4, 128, 0, stream>>>(agg1, invd, x, packW1, b1l, h, 1);
  }
  {
    long long tot = (long long)N_EDGES * (128 / 4);
    scatter_add_kernel<<<(unsigned)((tot + 255) / 256), 256, 0, stream>>>(h, ei, agg2, 128);
  }
  {
    int pairs = (N_NODES + 31) / 32;
    sage_linear_kernel<128><<<(pairs + 3) / 4, 128, 0, stream>>>(agg2, invd, h, packW2, b2l, z, 0);
  }
  {
    int pairs = (N_LBL + 31) / 32;      // 15625
    decoder_kernel<<<(pairs + 3) / 4, 128, 0, stream>>>(z, eli, packD1, db1, dw2, db2, out);
  }
}
